// TCAM_13675175870729
// MI455X (gfx1250) — matrix operand order repaired and checked by reading, not compiled, not run on hardware
//
#include <hip/hip_runtime.h>

// ---------------------------------------------------------------- types
typedef __bf16 bf16;
typedef __attribute__((ext_vector_type(16))) __bf16 v16bf;
typedef __attribute__((ext_vector_type(8)))  float  v8f;

#define B_  4
#define C_  512
#define H_  64
#define W_  64
#define H2  32
#define W2  32
#define L_  1024          // H2*W2
#define K1  4608          // C_*9
#define M2  8192          // C_*16

// d_out layout (floats): f_xt | mask | flow | grids
#define OUT_FXT   0
#define OUT_MASK  8388608           // 4*512*64*64
#define OUT_FLOW  8404992           // + 4*64*64
#define OUT_GRIDS 8413184           // + 4*2*32*32

union FragU { uint4 u[2]; v16bf v; };

// A-fragment (16x32 bf16, M rows): lane m = lane&15, hi = lane>=16.
// halves e0..7  -> K = kk + 8*hi + e      (16 contiguous bytes)
// halves e8..15 -> K = kk + 16 + 8*hi + e (16 contiguous bytes)
__device__ __forceinline__ v16bf load_fragA(const bf16* base, int ld, int row, int kk, int hi) {
  const bf16* p = base + (size_t)row * ld + kk + 8 * hi;
  FragU f;
  f.u[0] = *(const uint4*)(p);
  f.u[1] = *(const uint4*)(p + 16);
  return f.v;
}

// B-fragment (32x16 bf16, N columns from a row-major (N,K) source):
// lane n = lane&15, e0..7 -> K = kk + 8*hi + e, e8..15 -> K = kk + 16 + 8*hi + e
__device__ __forceinline__ v16bf load_fragB(const bf16* base, int ld, int row, int kk, int hi) {
  const bf16* p = base + (size_t)row * ld + kk + 8 * hi;
  FragU f;
  f.u[0] = *(const uint4*)(p);
  f.u[1] = *(const uint4*)(p + 16);
  return f.v;
}

__device__ __forceinline__ v8f wmma_bf16(v16bf a, v16bf b, v8f c) {
  return __builtin_amdgcn_wmma_f32_16x16x32_bf16(false, a, false, b, (short)0, c, false, false);
}

// ---------------------------------------------------------------- prep
__global__ void zero_kernel(float* __restrict__ p, size_t n) {
  size_t i = (size_t)blockIdx.x * blockDim.x + threadIdx.x;
  if (i < n) p[i] = 0.f;
}

__global__ void downsample2(const float* __restrict__ f_ct, const float* __restrict__ f_cs,
                            float* __restrict__ ct_d, float* __restrict__ cs_d) {
  int idx = blockIdx.x * blockDim.x + threadIdx.x;           // B*C*32*32
  if (idx >= B_ * C_ * H2 * W2) return;
  int x = idx & 31, y = (idx >> 5) & 31, bc = idx >> 10;
  size_t src = ((size_t)bc << 12) + ((size_t)(y * 2) << 6) + (size_t)(x * 2);
  ct_d[idx] = f_ct[src];
  cs_d[idx] = f_cs[src];
}

__global__ void sumsq_kernel(const float* __restrict__ cs_d, float* __restrict__ sq) {
  int idx = blockIdx.x * blockDim.x + threadIdx.x;           // B*1024
  if (idx >= B_ * L_) return;
  int b = idx >> 10, pos = idx & 1023;
  const float* p = cs_d + (((size_t)b * C_) << 10) + pos;
  float s = 0.f;
  for (int c = 0; c < C_; ++c) { float v = p[(size_t)c << 10]; s += v * v; }
  sq[idx] = s;
}

__global__ void invnorm_kernel(const float* __restrict__ sq, float* __restrict__ invn) {
  int idx = blockIdx.x * blockDim.x + threadIdx.x;           // B*1024
  if (idx >= B_ * L_) return;
  int b = idx >> 10, l = idx & 1023;
  int y = l >> 5, x = l & 31;
  float s = 0.f;
  for (int ky = 0; ky < 3; ++ky)
    for (int kx = 0; kx < 3; ++kx) {
      int yy = y + ky - 1, xx = x + kx - 1;
      if ((unsigned)yy < 32u && (unsigned)xx < 32u) s += sq[(b << 10) + (yy << 5) + xx];
    }
  invn[idx] = 1.f / fmaxf(sqrtf(s), 1e-4f);
}

// om_n rows (normalized 3x3xC patches of f_cs_d): A[b, l, k]  k=c*9+ky*3+kx
__global__ void build_A(const float* __restrict__ cs_d, const float* __restrict__ invn,
                        bf16* __restrict__ A) {
  size_t idx = (size_t)blockIdx.x * blockDim.x + threadIdx.x; // B*L*K1
  if (idx >= (size_t)B_ * L_ * K1) return;
  int k = (int)(idx % K1);
  size_t r = idx / K1;
  int l = (int)(r & 1023);
  int b = (int)(r >> 10);
  int c = k / 9, q = k % 9, ky = q / 3, kx = q % 3;
  int y = l >> 5, x = l & 31;
  int yy = y + ky - 1, xx = x + kx - 1;
  float v = 0.f;
  if ((unsigned)yy < 32u && (unsigned)xx < 32u)
    v = cs_d[(((size_t)b * C_ + c) << 10) + (yy << 5) + xx];
  A[idx] = (bf16)(v * invn[(b << 10) + l]);
}

// fq^T rows (3x3xC patches of f_ct_d): BT[b, p, k]
__global__ void build_BT(const float* __restrict__ ct_d, bf16* __restrict__ BT) {
  size_t idx = (size_t)blockIdx.x * blockDim.x + threadIdx.x; // B*L*K1
  if (idx >= (size_t)B_ * L_ * K1) return;
  int k = (int)(idx % K1);
  size_t r = idx / K1;
  int l = (int)(r & 1023);
  int b = (int)(r >> 10);
  int c = k / 9, q = k % 9, ky = q / 3, kx = q % 3;
  int y = l >> 5, x = l & 31;
  int yy = y + ky - 1, xx = x + kx - 1;
  float v = 0.f;
  if ((unsigned)yy < 32u && (unsigned)xx < 32u)
    v = ct_d[(((size_t)b * C_ + c) << 10) + (yy << 5) + xx];
  BT[idx] = (bf16)v;
}

// PhiT[b, m=(c*16+jy*4+jx), l=(li*32+lj)] = f_xs[b, c, 2li-1+jy, 2lj-1+jx]
__global__ void build_PhiT(const float* __restrict__ f_xs, bf16* __restrict__ PhiT) {
  size_t idx = (size_t)blockIdx.x * blockDim.x + threadIdx.x; // B*M2*L_
  if (idx >= (size_t)B_ * M2 * L_) return;
  int l = (int)(idx & 1023);
  size_t r = idx >> 10;
  int m = (int)(r & 8191);
  int b = (int)(r >> 13);
  int c = m >> 4, jy = (m >> 2) & 3, jx = m & 3;
  int Y = ((l >> 5) << 1) - 1 + jy;
  int X = ((l & 31) << 1) - 1 + jx;
  float v = 0.f;
  if ((unsigned)Y < 64u && (unsigned)X < 64u)
    v = f_xs[(((size_t)b * C_ + c) << 12) + (Y << 6) + X];
  PhiT[idx] = (bf16)v;
}

// ------------------------------------------------ GEMM-1: G[b,p,l] = scores^T
// D[p,l] = sum_k BT[p,k] * Aom[l,k]   (NT, both K-contiguous)
__global__ void gemm1_scores(const bf16* __restrict__ BT, const bf16* __restrict__ Aom,
                             float* __restrict__ G) {
  const int lane = threadIdx.x & 31;
  const int wave = blockIdx.x * (blockDim.x >> 5) + (threadIdx.x >> 5);
  const int b  = wave >> 10;                 // 32x32 tiles of 32x32 per batch
  const int t  = wave & 1023;
  const int p0 = (t >> 5) << 5;
  const int l0 = (t & 31) << 5;
  const int lm = lane & 15;
  const int hi = lane >> 4;

  const bf16* Ab = BT  + (size_t)b * L_ * K1;
  const bf16* Bb = Aom + (size_t)b * L_ * K1;

  v8f acc[2][2] = {};
  for (int kk = 0; kk < K1; kk += 32) {
    v16bf a0 = load_fragA(Ab, K1, p0 + lm,      kk, hi);
    v16bf a1 = load_fragA(Ab, K1, p0 + 16 + lm, kk, hi);
    v16bf b0 = load_fragB(Bb, K1, l0 + lm,      kk, hi);
    v16bf b1 = load_fragB(Bb, K1, l0 + 16 + lm, kk, hi);
    if (kk + 32 < K1) {
      __builtin_prefetch(Ab + (size_t)(p0 + lm) * K1 + kk + 32, 0, 0);
      __builtin_prefetch(Bb + (size_t)(l0 + lm) * K1 + kk + 32, 0, 0);
    }
    acc[0][0] = wmma_bf16(a0, b0, acc[0][0]);
    acc[0][1] = wmma_bf16(a0, b1, acc[0][1]);
    acc[1][0] = wmma_bf16(a1, b0, acc[1][0]);
    acc[1][1] = wmma_bf16(a1, b1, acc[1][1]);
  }
  float* Gb = G + (size_t)b * L_ * L_;
#pragma unroll
  for (int i = 0; i < 2; ++i)
#pragma unroll
    for (int r = 0; r < 8; ++r) {
      int mr = p0 + i * 16 + r + 8 * hi;
      Gb[(size_t)mr * L_ + l0 + lm]      = acc[i][0][r];
      Gb[(size_t)mr * L_ + l0 + 16 + lm] = acc[i][1][r];
    }
}

// ---------------------------------------- softmax over l per (b,p) + argmax
__global__ void softmax_attn(const float* __restrict__ G, bf16* __restrict__ AttnT,
                             float* __restrict__ flow, float* __restrict__ grids) {
  __shared__ float sval[256];
  __shared__ int   sidx[256];
  __shared__ float ssum[256];
  const int tid = threadIdx.x;
  const int bp  = blockIdx.x;               // b*1024 + p
  const float* row = G + (size_t)bp * L_;

  float best = -1e30f; int bidx = 0;
  float vloc[4];
#pragma unroll
  for (int k = 0; k < 4; ++k) {
    int l = tid + k * 256;
    float v = row[l];
    vloc[k] = v;
    if (v > best) { best = v; bidx = l; }
  }
  sval[tid] = best; sidx[tid] = bidx;
  __syncthreads();
  for (int s = 128; s > 0; s >>= 1) {
    if (tid < s) {
      float v2 = sval[tid + s]; int i2 = sidx[tid + s];
      if (v2 > sval[tid] || (v2 == sval[tid] && i2 < sidx[tid])) { sval[tid] = v2; sidx[tid] = i2; }
    }
    __syncthreads();
  }
  const float mx = sval[0];
  const int amax = sidx[0];

  float s = 0.f;
#pragma unroll
  for (int k = 0; k < 4; ++k) s += expf(10.f * (vloc[k] - mx));
  ssum[tid] = s;
  __syncthreads();
  for (int sr = 128; sr > 0; sr >>= 1) {
    if (tid < sr) ssum[tid] += ssum[tid + sr];
    __syncthreads();
  }
  const float inv = 1.f / ssum[0];

  bf16* arow = AttnT + (size_t)bp * L_;
#pragma unroll
  for (int k = 0; k < 4; ++k) {
    int l = tid + k * 256;
    arow[l] = (bf16)(expf(10.f * (vloc[k] - mx)) * inv);
  }
  if (tid == 0) {
    int b = bp >> 10, p = bp & 1023;
    int xo = amax & 31, yo = amax >> 5;
    int i = p >> 5, j = p & 31;
    flow [((size_t)(b * 2 + 0) << 10) + p] = (float)(xo - i);
    flow [((size_t)(b * 2 + 1) << 10) + p] = (float)(yo - j);
    grids[((size_t)(b * 2 + 0) << 10) + p] = (float)xo;
    grids[((size_t)(b * 2 + 1) << 10) + p] = (float)yo;
  }
}

// -------------------- GEMM-2 + fused col2im scatter (transposed conv paste)
// col[m,p] = sum_l PhiT[m,l] * AttnT[p,l]; scatter to f_xt at Y=2i-1+jy, X=2j-1+jx
__global__ void gemm2_paste(const bf16* __restrict__ PhiT, const bf16* __restrict__ AttnT,
                            float* __restrict__ fxt) {
  const int lane = threadIdx.x & 31;
  const int wave = blockIdx.x * (blockDim.x >> 5) + (threadIdx.x >> 5);
  const int b  = wave >> 13;                 // 256x32 tiles of 32x32 per batch
  const int t  = wave & 8191;
  const int m0 = (t >> 5) << 5;
  const int p0 = (t & 31) << 5;
  const int lm = lane & 15;
  const int hi = lane >> 4;

  const bf16* Ab = PhiT  + (size_t)b * M2 * L_;
  const bf16* Bb = AttnT + (size_t)b * L_ * L_;

  v8f acc[2][2] = {};
  for (int kk = 0; kk < L_; kk += 32) {
    v16bf a0 = load_fragA(Ab, L_, m0 + lm,      kk, hi);
    v16bf a1 = load_fragA(Ab, L_, m0 + 16 + lm, kk, hi);
    v16bf b0 = load_fragB(Bb, L_, p0 + lm,      kk, hi);
    v16bf b1 = load_fragB(Bb, L_, p0 + 16 + lm, kk, hi);
    if (kk + 32 < L_) {
      __builtin_prefetch(Ab + (size_t)(m0 + lm) * L_ + kk + 32, 0, 0);
      __builtin_prefetch(Bb + (size_t)(p0 + lm) * L_ + kk + 32, 0, 0);
    }
    acc[0][0] = wmma_bf16(a0, b0, acc[0][0]);
    acc[0][1] = wmma_bf16(a0, b1, acc[0][1]);
    acc[1][0] = wmma_bf16(a1, b0, acc[1][0]);
    acc[1][1] = wmma_bf16(a1, b1, acc[1][1]);
  }

  float* outb = fxt + (size_t)b * C_ * H_ * W_;
#pragma unroll
  for (int i = 0; i < 2; ++i)
#pragma unroll
    for (int j = 0; j < 2; ++j)
#pragma unroll
      for (int r = 0; r < 8; ++r) {
        int m = m0 + i * 16 + r + 8 * hi;
        int p = p0 + j * 16 + lm;
        int c  = m >> 4;
        int jy = (m >> 2) & 3;
        int jx = m & 3;
        int Y = ((p >> 5) << 1) - 1 + jy;
        int X = ((p & 31) << 1) - 1 + jx;
        if ((unsigned)Y < 64u && (unsigned)X < 64u)
          atomicAdd(&outb[((size_t)c << 12) + (Y << 6) + X], acc[i][j][r] * 0.25f);
      }
}

// ---------------------------------------------------------------- mask conv
__global__ void mask_conv(const float* __restrict__ fxt, const float* __restrict__ w,
                          const float* __restrict__ bias, float* __restrict__ mask) {
  __shared__ float red[256];
  const int tid = threadIdx.x;
  const int bp  = blockIdx.x;               // b*4096 + pos
  const int b = bp >> 12, pos = bp & 4095;
  const int y = pos >> 6, x = pos & 63;
  float acc = 0.f;
  for (int c = tid; c < C_; c += 256) {
    const float* fp = fxt + (((size_t)b * C_ + c) << 12);
    const float* wp = w + c * 9;
#pragma unroll
    for (int ky = 0; ky < 3; ++ky) {
      int yy = y + ky - 1;
      if ((unsigned)yy >= 64u) continue;
#pragma unroll
      for (int kx = 0; kx < 3; ++kx) {
        int xx = x + kx - 1;
        if ((unsigned)xx >= 64u) continue;
        acc += fp[(yy << 6) + xx] * wp[ky * 3 + kx];
      }
    }
  }
  red[tid] = acc;
  __syncthreads();
  for (int s = 128; s > 0; s >>= 1) {
    if (tid < s) red[tid] += red[tid + s];
    __syncthreads();
  }
  if (tid == 0) {
    float v = red[0] + bias[0];
    mask[bp] = 1.f / (1.f + expf(-v));
  }
}

// ---------------------------------------------------------------- launch
extern "C" void kernel_launch(void* const* d_in, const int* in_sizes, int n_in,
                              void* d_out, int out_size, void* d_ws, size_t ws_size,
                              hipStream_t stream) {
  (void)in_sizes; (void)n_in; (void)ws_size;
  const float* f_ct = (const float*)d_in[0];
  const float* f_cs = (const float*)d_in[1];
  const float* f_xs = (const float*)d_in[2];
  const float* mw   = (const float*)d_in[3];
  const float* mb   = (const float*)d_in[4];
  float* out = (float*)d_out;

  char* ws = (char*)d_ws;
  size_t o = 0;
  auto alloc = [&](size_t bytes) -> char* {
    char* r = ws + o;
    o = (o + bytes + 255) & ~(size_t)255;
    return r;
  };
  float* ct_d  = (float*)alloc((size_t)B_ * C_ * H2 * W2 * 4);   //   8.4 MB
  float* cs_d  = (float*)alloc((size_t)B_ * C_ * H2 * W2 * 4);   //   8.4 MB
  float* sq    = (float*)alloc((size_t)B_ * L_ * 4);
  float* invn  = (float*)alloc((size_t)B_ * L_ * 4);
  bf16*  Aom   = (bf16*) alloc((size_t)B_ * L_ * K1 * 2);        //  37.7 MB
  bf16*  BTfq  = (bf16*) alloc((size_t)B_ * L_ * K1 * 2);        //  37.7 MB
  float* G     = (float*)alloc((size_t)B_ * L_ * L_ * 4);        //  16.8 MB
  bf16*  AttnT = (bf16*) alloc((size_t)B_ * L_ * L_ * 2);        //   8.4 MB
  bf16*  PhiT  = (bf16*) alloc((size_t)B_ * M2 * L_ * 2);        //  67.1 MB

  const size_t nout = (size_t)out_size;
  zero_kernel<<<(unsigned)((nout + 255) / 256), 256, 0, stream>>>(out, nout);
  downsample2<<<(B_ * C_ * H2 * W2 + 255) / 256, 256, 0, stream>>>(f_ct, f_cs, ct_d, cs_d);
  sumsq_kernel<<<(B_ * L_ + 255) / 256, 256, 0, stream>>>(cs_d, sq);
  invnorm_kernel<<<(B_ * L_ + 255) / 256, 256, 0, stream>>>(sq, invn);
  build_A <<<(unsigned)(((size_t)B_ * L_ * K1 + 255) / 256), 256, 0, stream>>>(cs_d, invn, Aom);
  build_BT<<<(unsigned)(((size_t)B_ * L_ * K1 + 255) / 256), 256, 0, stream>>>(ct_d, BTfq);
  // 4096 wave-tiles (32x32 each), 8 waves per 256-thread block
  gemm1_scores<<<512, 256, 0, stream>>>(BTfq, Aom, G);
  softmax_attn<<<B_ * L_, 256, 0, stream>>>(G, AttnT, out + OUT_FLOW, out + OUT_GRIDS);
  build_PhiT<<<(unsigned)(((size_t)B_ * M2 * L_ + 255) / 256), 256, 0, stream>>>(f_xs, PhiT);
  // 32768 wave-tiles, 8 waves per block
  gemm2_paste<<<4096, 256, 0, stream>>>(PhiT, AttnT, out + OUT_FXT);
  mask_conv<<<B_ * H_ * W_, 256, 0, stream>>>(out + OUT_FXT, mw, mb, out + OUT_MASK);
}
